// Attention_39934605918652
// MI455X (gfx1250) — matrix operand order repaired and checked by reading, not compiled, not run on hardware
//
#include <hip/hip_runtime.h>
#include <math.h>
#include <stdint.h>

typedef __attribute__((ext_vector_type(16))) _Float16 v16h;
typedef __attribute__((ext_vector_type(8)))  _Float16 v8h;
typedef __attribute__((ext_vector_type(8)))  float    v8f;
typedef __attribute__((ext_vector_type(4)))  float    v4f;

#define FEAT 512
#define HID  1024
#define BB   128
#define NN   256

#if __has_builtin(__builtin_amdgcn_s_wait_asynccnt)
#define WAIT_ASYNC(n) __builtin_amdgcn_s_wait_asynccnt((unsigned short)(n))
#else
#define WAIT_ASYNC(n) asm volatile("s_wait_asynccnt %0" ::"i"(n) : "memory")
#endif

// Async DMA: global (16B/lane) -> LDS, tracked by ASYNCcnt. LDS address is the
// low 32 bits of the flat pointer.
__device__ __forceinline__ void async_b128(const void* g, void* l) {
    unsigned long long ga = (unsigned long long)(uintptr_t)g;
    unsigned           la = (unsigned)(uintptr_t)l;
    asm volatile("global_load_async_to_lds_b128 %0, %1, off" ::"v"(la), "v"(ga) : "memory");
}

// CDNA5 hardware transcendental tanh (single TRANS-unit instruction).
__device__ __forceinline__ float fast_tanh(float x) {
#if __has_builtin(__builtin_amdgcn_tanhf)
    return __builtin_amdgcn_tanhf(x);
#else
    float r;
    asm("v_tanh_f32 %0, %1" : "=v"(r) : "v"(x));
    return r;
#endif
}

// ---------------- fragment loaders ----------------

// A-matrix 16x32 f16 fragment from f16 row-major global.
// Lanes 0-15: K = k0+{0..7,16..23}; lanes 16-31: K = k0+{8..15,24..31}.
__device__ __forceinline__ v16h a_frag_f16(const _Float16* __restrict__ rowp, int k0, int lhi) {
    const _Float16* r = rowp + k0 + (lhi ? 8 : 0);
    v8h lo = *(const v8h*)(r);
    v8h hi = *(const v8h*)(r + 16);
    v16h o;
#pragma unroll
    for (int i = 0; i < 8; ++i) { o[i] = lo[i]; o[8 + i] = hi[i]; }
    return o;
}

// A fragment converting from f32 row-major source (used for S).
__device__ __forceinline__ v16h a_frag_f32(const float* __restrict__ rowp, int k0, int lhi) {
    const float* r = rowp + k0 + (lhi ? 8 : 0);
    v4f x0 = *(const v4f*)(r + 0);
    v4f x1 = *(const v4f*)(r + 4);
    v4f y0 = *(const v4f*)(r + 16);
    v4f y1 = *(const v4f*)(r + 20);
    v16h o;
#pragma unroll
    for (int i = 0; i < 4; ++i) {
        o[i]      = (_Float16)x0[i];
        o[4 + i]  = (_Float16)x1[i];
        o[8 + i]  = (_Float16)y0[i];
        o[12 + i] = (_Float16)y1[i];
    }
    return o;
}

// B-matrix 32x16 f16 fragment (rowp = &B[n_col*ld]).
// Lanes 0-15: K = k0+{0..7,16..23}; lanes 16-31: K = k0+{8..15,24..31}.
__device__ __forceinline__ v16h b_frag_f16(const _Float16* __restrict__ rowp, int k0, int lhi) {
    const _Float16* r = rowp + k0 + (lhi ? 8 : 0);
    v8h lo = *(const v8h*)(r);
    v8h hi = *(const v8h*)(r + 16);
    v16h o;
#pragma unroll
    for (int i = 0; i < 8; ++i) { o[i] = lo[i]; o[8 + i] = hi[i]; }
    return o;
}

// A fragment from LDS (f16, XOR-swizzled 16B granules, row stride 512 halfs).
__device__ __forceinline__ v16h a_frag_lds(const _Float16* As, int row, int k0, int lhi, int sw) {
    int g0 = (k0 >> 3) + lhi;
    const _Float16* p0 = As + row * 512 + ((g0) ^ sw) * 8;
    const _Float16* p1 = As + row * 512 + ((g0 + 2) ^ sw) * 8;
    v8h lo = *(const v8h*)p0;
    v8h hi = *(const v8h*)p1;
    v16h o;
#pragma unroll
    for (int i = 0; i < 8; ++i) { o[i] = lo[i]; o[8 + i] = hi[i]; }
    return o;
}

#define WMMA_F16(a, b, c) \
    __builtin_amdgcn_wmma_f32_16x16x32_f16(false, (a), false, (b), (short)0, (c), false, false)

// ---------------- conversion kernels (NT loads: single-use f32 streams) ----------------

__global__ __launch_bounds__(256) void convert_x_kernel(const float* __restrict__ x,
                                                        _Float16* __restrict__ xh) {
    size_t idx = (size_t)blockIdx.x * 256 + threadIdx.x;   // one 8-element group each
    const v4f* s = (const v4f*)x + idx * 2;
    v4f a = __builtin_nontemporal_load(s);
    v4f b = __builtin_nontemporal_load(s + 1);
    v8h o;
#pragma unroll
    for (int i = 0; i < 4; ++i) { o[i] = (_Float16)a[i]; o[4 + i] = (_Float16)b[i]; }
    *(v8h*)(xh + idx * 8) = o;
}

__global__ __launch_bounds__(256) void convert_w1_kernel(const float* __restrict__ W1,
                                                         _Float16* __restrict__ W1a,
                                                         _Float16* __restrict__ W1b) {
    int idx = blockIdx.x * 256 + threadIdx.x;     // over 1024*1024/4 float4s
    v4f v = __builtin_nontemporal_load((const v4f*)W1 + idx);
    int h  = idx >> 8;
    int c4 = idx & 255;
    _Float16* dst = (c4 < 128) ? (W1a + h * FEAT + c4 * 4)
                               : (W1b + h * FEAT + (c4 - 128) * 4);
#pragma unroll
    for (int i = 0; i < 4; ++i) dst[i] = (_Float16)v[i];
}

__global__ __launch_bounds__(256) void convert_f16_kernel(const float* __restrict__ src,
                                                          _Float16* __restrict__ dst, int n4) {
    int idx = blockIdx.x * 256 + threadIdx.x;
    if (idx < n4) {
        v4f v = __builtin_nontemporal_load((const v4f*)src + idx);
#pragma unroll
        for (int i = 0; i < 4; ++i) dst[idx * 4 + i] = (_Float16)v[i];
    }
}

// ---------------- h0 = x0 @ W1a^T  (128 x 1024, K=512) ----------------

__global__ __launch_bounds__(128) void h0_kernel(const _Float16* __restrict__ xh,
                                                 const _Float16* __restrict__ W1a,
                                                 float* __restrict__ h0out) {
    const int tid = threadIdx.x, lane = tid & 31, wave = tid >> 5;
    const int l16 = lane & 15, lhi = lane >> 4;
    const int hbase = blockIdx.x * 128 + wave * 32;
    const int mbase = blockIdx.y * 16;

    const _Float16* Arow = xh + (size_t)(mbase + l16) * NN * FEAT;  // x0 rows
    const _Float16* Bp0  = W1a + (size_t)(hbase + l16) * FEAT;
    const _Float16* Bp1  = Bp0 + (size_t)16 * FEAT;

    v8f c0 = {}, c1 = {};
    for (int k0 = 0; k0 < FEAT; k0 += 32) {
        v16h a  = a_frag_f16(Arow, k0, lhi);
        c0 = WMMA_F16(a, b_frag_f16(Bp0, k0, lhi), c0);
        c1 = WMMA_F16(a, b_frag_f16(Bp1, k0, lhi), c1);
    }
#pragma unroll
    for (int v = 0; v < 8; ++v) {
        int m = mbase + v + lhi * 8;
        h0out[m * HID + hbase + l16]      = c0[v];
        h0out[m * HID + hbase + 16 + l16] = c1[v];
    }
}

// ---------------- S[b,h] = sum_{n=1..255} tanh(h0[b,h] + (x[b,n,:]@W1b^T)[h]) ----------------

__global__ __launch_bounds__(256) void interact_kernel(const _Float16* __restrict__ xh,
                                                       const _Float16* __restrict__ W1b,
                                                       const float* __restrict__ h0,
                                                       float* __restrict__ S) {
    extern __shared__ _Float16 As[];              // 2 x (64 x 512) halfs = 128 KB dynamic LDS

    const int b = blockIdx.y;
    const int htile = blockIdx.x;                 // 0..3, 256 h-columns per block
    const int tid = threadIdx.x;
    const int lane = tid & 31, wave = tid >> 5;
    const int l16 = lane & 15, lhi = lane >> 4;

    const int hbase = htile * 256 + wave * 32;
    const _Float16* Bp0 = W1b + (size_t)(hbase + l16) * FEAT;
    const _Float16* Bp1 = Bp0 + (size_t)16 * FEAT;

    const float h0c0 = h0[b * HID + hbase + l16];
    const float h0c1 = h0[b * HID + hbase + 16 + l16];

    float s0 = 0.f, s1 = 0.f;
    const _Float16* xbh = xh + (size_t)b * NN * FEAT;

    // Issue the async DMA for one 64-row chunk into LDS buffer `buf`,
    // applying the XOR granule swizzle in the per-lane LDS dest address.
    auto issue = [&](int chunk, int buf) {
#pragma unroll
        for (int i = 0; i < 16; ++i) {
            int g_lin = tid + i * 256;            // 0..4095 16B-granules
            int row   = g_lin >> 6;               // 64 granules per row
            int gcol  = g_lin & 63;
            const _Float16* src = xbh + (size_t)(chunk * 64 + row) * FEAT + gcol * 8;
            _Float16* dst = &As[buf * (64 * 512) + row * 512 + (gcol ^ (row & 15)) * 8];
            async_b128(src, dst);
        }
    };

    issue(0, 0);

    for (int chunk = 0; chunk < 4; ++chunk) {
        if (chunk < 3) issue(chunk + 1, (chunk + 1) & 1);   // prefetch next chunk
        if (chunk < 3) { WAIT_ASYNC(16); } else { WAIT_ASYNC(0); }
        __syncthreads();                                     // all waves' DMA now visible

        const _Float16* Ab = As + (chunk & 1) * (64 * 512);

        v8f acc[4][2];
#pragma unroll
        for (int mf = 0; mf < 4; ++mf) { acc[mf][0] = (v8f){}; acc[mf][1] = (v8f){}; }

        for (int k0 = 0; k0 < FEAT; k0 += 32) {
            v16h b0 = b_frag_f16(Bp0, k0, lhi);
            v16h b1 = b_frag_f16(Bp1, k0, lhi);
#pragma unroll
            for (int mf = 0; mf < 4; ++mf) {
                v16h a = a_frag_lds(Ab, mf * 16 + l16, k0, lhi, l16);
                acc[mf][0] = WMMA_F16(a, b0, acc[mf][0]);
                acc[mf][1] = WMMA_F16(a, b1, acc[mf][1]);
            }
        }

        // hardware tanh + reduction over the n (row) dimension, masking n==0
#pragma unroll
        for (int mf = 0; mf < 4; ++mf) {
#pragma unroll
            for (int v = 0; v < 8; ++v) {
                int n = chunk * 64 + mf * 16 + v + lhi * 8;
                if (n != 0) {
                    s0 += fast_tanh(acc[mf][0][v] + h0c0);
                    s1 += fast_tanh(acc[mf][1][v] + h0c1);
                }
            }
        }
        __syncthreads();    // reads of this buffer done before it is re-filled
    }

    s0 += __shfl_xor(s0, 16, 32);
    s1 += __shfl_xor(s1, 16, 32);
    if (lhi == 0) {
        S[b * HID + hbase + l16]      = s0;
        S[b * HID + hbase + 16 + l16] = s1;
    }
}

// ---------------- res = x0 @ W0^T + S @ W2^T  (128 x 512) ----------------

__global__ __launch_bounds__(128) void out_kernel(const _Float16* __restrict__ xh,
                                                  const float* __restrict__ S,
                                                  const _Float16* __restrict__ W0h,
                                                  const _Float16* __restrict__ W2h,
                                                  float* __restrict__ res) {
    const int tid = threadIdx.x, lane = tid & 31, wave = tid >> 5;
    const int l16 = lane & 15, lhi = lane >> 4;
    const int fbase = blockIdx.x * 128 + wave * 32;
    const int mbase = blockIdx.y * 16;

    const _Float16* A1row = xh + (size_t)(mbase + l16) * NN * FEAT;   // x0 rows
    const float*    A2row = S + (size_t)(mbase + l16) * HID;

    const _Float16* B10 = W0h + (size_t)(fbase + l16) * FEAT;
    const _Float16* B11 = B10 + (size_t)16 * FEAT;
    const _Float16* B20 = W2h + (size_t)(fbase + l16) * HID;
    const _Float16* B21 = B20 + (size_t)16 * HID;

    v8f c0 = {}, c1 = {};
    for (int k0 = 0; k0 < FEAT; k0 += 32) {
        v16h a = a_frag_f16(A1row, k0, lhi);
        c0 = WMMA_F16(a, b_frag_f16(B10, k0, lhi), c0);
        c1 = WMMA_F16(a, b_frag_f16(B11, k0, lhi), c1);
    }
    for (int k0 = 0; k0 < HID; k0 += 32) {
        v16h a = a_frag_f32(A2row, k0, lhi);
        c0 = WMMA_F16(a, b_frag_f16(B20, k0, lhi), c0);
        c1 = WMMA_F16(a, b_frag_f16(B21, k0, lhi), c1);
    }
#pragma unroll
    for (int v = 0; v < 8; ++v) {
        int m = mbase + v + lhi * 8;
        res[m * FEAT + fbase + l16]      = c0[v];
        res[m * FEAT + fbase + 16 + l16] = c1[v];
    }
}

// ---------------- launch ----------------

extern "C" void kernel_launch(void* const* d_in, const int* in_sizes, int n_in,
                              void* d_out, int out_size, void* d_ws, size_t ws_size,
                              hipStream_t stream) {
    const float* x  = (const float*)d_in[0];
    const float* W1 = (const float*)d_in[1];
    const float* W2 = (const float*)d_in[2];
    const float* W0 = (const float*)d_in[3];
    float* out = (float*)d_out;

    char* ws = (char*)d_ws;
    float*    h0    = (float*)(ws + 0);                 // 128*1024 f32   (512 KB)
    float*    S     = (float*)(ws + 524288);            // 128*1024 f32   (512 KB)
    _Float16* W1a_h = (_Float16*)(ws + 1048576);        // 1024*512 f16   (1 MB)
    _Float16* W1b_h = (_Float16*)(ws + 2097152);        // 1024*512 f16   (1 MB)
    _Float16* W2_h  = (_Float16*)(ws + 3145728);        // 512*1024 f16   (1 MB)
    _Float16* W0_h  = (_Float16*)(ws + 4194304);        // 512*512 f16    (512 KB)
    _Float16* x_h   = (_Float16*)(ws + 4718592);        // 128*256*512 f16 (32 MB)

    convert_x_kernel<<<(BB * NN * FEAT) / 8 / 256, 256, 0, stream>>>(x, x_h);
    convert_w1_kernel<<<1024, 256, 0, stream>>>(W1, W1a_h, W1b_h);
    convert_f16_kernel<<<512, 256, 0, stream>>>(W2, W2_h, (FEAT * HID) / 4);
    convert_f16_kernel<<<256, 256, 0, stream>>>(W0, W0_h, (FEAT * FEAT) / 4);

    h0_kernel<<<dim3(HID / 128, BB / 16), 128, 0, stream>>>(x_h, W1a_h, h0);
    interact_kernel<<<dim3(4, BB), 256, 2 * 64 * 512 * sizeof(_Float16), stream>>>(x_h, W1b_h, h0, S);
    out_kernel<<<dim3(FEAT / 128, BB / 16), 128, 0, stream>>>(x_h, S, W0_h, W2_h, out);
}
